// NERRelationModel_63917703299394
// MI455X (gfx1250) — compile-verified, matrix operand order checked by reading, not run on hardware
//
#include <hip/hip_runtime.h>

typedef _Float16 f16;
typedef __attribute__((ext_vector_type(16))) _Float16 v16h;
typedef __attribute__((ext_vector_type(8)))  float    v8f;

#define LDS_STRIDE 48  // f16 elements; 96B rows keep 16B alignment for b128 LDS ops

#if defined(__gfx1250__) && __has_builtin(__builtin_amdgcn_global_load_async_to_lds_b128) && __has_builtin(__builtin_amdgcn_s_wait_asynccnt)
#define HAVE_ASYNC_LDS 1
typedef int v4i __attribute__((vector_size(16)));
typedef __attribute__((address_space(1))) v4i* gptr_v4i;
typedef __attribute__((address_space(3))) v4i* lptr_v4i;
#else
#define HAVE_ASYNC_LDS 0
#endif

// ---------------------------------------------------------------------------
// WMMA helpers. Fragment layout per CDNA5 ISA (16-bit A 16x32):
// lane = half*16 + idx; element e holds k = half*8 + e (e<8) or 16+half*8+(e-8).
// B assumed mirrored (n = idx, same k pattern). C/D: VGPR v -> M = half*8+v, N = idx.
// ---------------------------------------------------------------------------
__device__ __forceinline__ v16h frag_ld(const f16* p, int half) {
  v16h r;
  const f16* p0 = p + half * 8;
#pragma unroll
  for (int e = 0; e < 8; ++e) r[e] = p0[e];
#pragma unroll
  for (int e = 0; e < 8; ++e) r[8 + e] = p0[16 + e];
  return r;
}

__device__ __forceinline__ v8f wmma16(v16h a, v16h b, v8f c) {
  return __builtin_amdgcn_wmma_f32_16x16x32_f16(false, a, false, b, (short)0, c,
                                                false, false);
}

// ---------------------------------------------------------------------------
// Generic tiled GEMM: C[M,N] = act(A[M,K] @ B[K,N] + bias)
// A,B f16 row-major. 256 threads = 8 waves; block tile 128x128; wave tile 32x64.
// M % 128 == 0, K % 32 == 0 required (true for all call sites). N guarded only
// in the (rare) partial-tile path so the common path stays fully vectorized.
// ---------------------------------------------------------------------------
__global__ void __launch_bounds__(256)
gemm_wmma(const f16* __restrict__ A, const f16* __restrict__ Bw,
          const float* __restrict__ bias, float* __restrict__ Cf,
          f16* __restrict__ Ch, int M, int N, int K, int relu) {
  __shared__ f16 As[128 * LDS_STRIDE];
  __shared__ f16 Bs[128 * LDS_STRIDE];
  const int tid = threadIdx.x;
  const int n0 = blockIdx.x * 128, m0 = blockIdx.y * 128;
  const int w = tid >> 5, lane = tid & 31, half = lane >> 4, idx = lane & 15;
  const int wm = w & 3, wn = w >> 2;
  const bool full_n = (n0 + 128 <= N);

  v8f acc[2][4];
#pragma unroll
  for (int i = 0; i < 2; ++i)
#pragma unroll
    for (int j = 0; j < 4; ++j) { v8f z = {}; acc[i][j] = z; }

  // A staging addresses (constant across k loop except kk0)
  const int ar = tid >> 1, apart = tid & 1;
  f16* adst = &As[ar * LDS_STRIDE + apart * 16];

  for (int kk0 = 0; kk0 < K; kk0 += 32) {
    __syncthreads();
    {  // stage A tile: 128 rows x 32 k, 16 contiguous f16 (32B) per thread
      const f16* src = A + (size_t)(m0 + ar) * K + kk0 + apart * 16;
#if HAVE_ASYNC_LDS
      __builtin_amdgcn_global_load_async_to_lds_b128(
          (gptr_v4i)(const void*)src, (lptr_v4i)(void*)adst, 0, 0);
      __builtin_amdgcn_global_load_async_to_lds_b128(
          (gptr_v4i)(const void*)(src + 8), (lptr_v4i)(void*)(adst + 8), 0, 0);
#else
#pragma unroll
      for (int i = 0; i < 16; ++i) adst[i] = src[i];
#endif
    }
    {  // stage B tile transposed into Bs[n][kk]
      int kk = tid & 31, ng = tid >> 5;
      const f16* src = Bw + (size_t)(kk0 + kk) * N + n0 + ng * 16;
      if (full_n) {
#pragma unroll
        for (int i = 0; i < 16; ++i) Bs[(ng * 16 + i) * LDS_STRIDE + kk] = src[i];
      } else {
#pragma unroll
        for (int i = 0; i < 16; ++i) {
          int n = ng * 16 + i;
          Bs[n * LDS_STRIDE + kk] = (n0 + n < N) ? src[i] : (f16)0.f;
        }
      }
    }
#if HAVE_ASYNC_LDS
    __builtin_amdgcn_s_wait_asynccnt(0);
#endif
    __syncthreads();
    v16h bf[4];
#pragma unroll
    for (int nt = 0; nt < 4; ++nt)
      bf[nt] = frag_ld(&Bs[(wn * 64 + nt * 16 + idx) * LDS_STRIDE], half);
#pragma unroll
    for (int mt = 0; mt < 2; ++mt) {
      v16h af = frag_ld(&As[(wm * 32 + mt * 16 + idx) * LDS_STRIDE], half);
#pragma unroll
      for (int nt = 0; nt < 4; ++nt) acc[mt][nt] = wmma16(af, bf[nt], acc[mt][nt]);
    }
  }

#pragma unroll
  for (int mt = 0; mt < 2; ++mt)
#pragma unroll
    for (int nt = 0; nt < 4; ++nt) {
      int c = n0 + wn * 64 + nt * 16 + idx;
      if (c >= N) continue;
      float bv = bias ? bias[c] : 0.f;
#pragma unroll
      for (int v = 0; v < 8; ++v) {
        int r = m0 + wm * 32 + mt * 16 + half * 8 + v;
        float val = acc[mt][nt][v] + bv;
        if (relu) val = fmaxf(val, 0.f);
        if (Cf) Cf[(size_t)r * N + c] = val;
        if (Ch) Ch[(size_t)r * N + c] = (f16)val;
      }
    }
}

// ---------------------------------------------------------------------------
// f32 -> f16 conversion, 4 elements per thread
// ---------------------------------------------------------------------------
__global__ void f32_to_f16(const float* __restrict__ in, f16* __restrict__ out,
                           size_t n) {
  size_t base = ((size_t)blockIdx.x * 256 + threadIdx.x) * 4;
  if (base + 4 <= n) {
#pragma unroll
    for (int i = 0; i < 4; ++i) out[base + i] = (f16)in[base + i];
  } else {
    for (size_t i = base; i < n; ++i) out[i] = (f16)in[i];
  }
}

// ---------------------------------------------------------------------------
// Entity span mean pooling: ent[b,e,:] = mean(seq[b, s0..s0+L, :])
// ---------------------------------------------------------------------------
__global__ void ent_pool(const float* __restrict__ seq, const int* __restrict__ starts,
                         const int* __restrict__ lens, float* __restrict__ ent,
                         f16* __restrict__ enth) {
  int be = blockIdx.x;  // b*32 + e
  int b = be >> 5;
  int s0 = starts[be], L = lens[be];
  float inv = 1.f / (float)(L + 1);
  for (int h = threadIdx.x; h < 768; h += 256) {
    float acc = 0.f;
    for (int s = s0; s <= s0 + L; ++s)
      acc += seq[((size_t)b * 1024 + s) * 768 + h];
    float v = acc * inv;
    ent[(size_t)be * 768 + h] = v;
    enth[(size_t)be * 768 + h] = (f16)v;
  }
}

// ---------------------------------------------------------------------------
// Attention scores: Sc[b,e,s] = (q[b,e,:] . k[b,s,:]) / sqrt(768), masked.
// ---------------------------------------------------------------------------
__global__ void __launch_bounds__(256)
attn_scores(const f16* __restrict__ Q, const f16* __restrict__ Kmat,
            const int* __restrict__ mask, float* __restrict__ Sc) {
  __shared__ f16 Qs[32 * 768];
  const int tid = threadIdx.x;
  const int b = blockIdx.y, c0 = blockIdx.x * 128;
  // copy 32x768 f16 in 8-element (16B) chunks
  for (int ch = tid; ch < 32 * 768 / 8; ch += 256) {
    const f16* src = Q + (size_t)b * 32 * 768 + ch * 8;
    f16* dst = &Qs[ch * 8];
#pragma unroll
    for (int i = 0; i < 8; ++i) dst[i] = src[i];
  }
  __syncthreads();
  const int w = tid >> 5, lane = tid & 31, half = lane >> 4, idx = lane & 15;
  const int nt0 = c0 + w * 16;
  v8f a0 = {}, a1 = {};
  for (int kk0 = 0; kk0 < 768; kk0 += 32) {
    v16h bf = frag_ld(&Kmat[((size_t)b * 1024 + nt0 + idx) * 768 + kk0], half);
    v16h f0 = frag_ld(&Qs[idx * 768 + kk0], half);
    v16h f1 = frag_ld(&Qs[(16 + idx) * 768 + kk0], half);
    a0 = wmma16(f0, bf, a0);
    a1 = wmma16(f1, bf, a1);
  }
  const float scale = 0.03608439182435161f;  // 1/sqrt(768)
  int col = nt0 + idx;
  int mk = mask[b * 1024 + col];
#pragma unroll
  for (int v = 0; v < 8; ++v) {
    int m = half * 8 + v;
    float s0 = a0[v] * scale, s1 = a1[v] * scale;
    if (mk == 0) { s0 = -1e9f; s1 = -1e9f; }
    Sc[((size_t)(b * 32 + m)) * 1024 + col] = s0;
    Sc[((size_t)(b * 32 + 16 + m)) * 1024 + col] = s1;
  }
}

// ---------------------------------------------------------------------------
// In-place softmax over rows of length 1024; one wave per row.
// ---------------------------------------------------------------------------
__global__ void __launch_bounds__(256) softmax_rows(float* __restrict__ S) {
  int w = threadIdx.x >> 5, lane = threadIdx.x & 31;
  size_t row = (size_t)blockIdx.x * 8 + w;
  float* p = S + row * 1024;
  float vals[32];
  float mx = -1e30f;
#pragma unroll
  for (int i = 0; i < 32; ++i) { vals[i] = p[i * 32 + lane]; mx = fmaxf(mx, vals[i]); }
#pragma unroll
  for (int m = 16; m > 0; m >>= 1) mx = fmaxf(mx, __shfl_xor(mx, m, 32));
  float sum = 0.f;
#pragma unroll
  for (int i = 0; i < 32; ++i) { vals[i] = __expf(vals[i] - mx); sum += vals[i]; }
#pragma unroll
  for (int m = 16; m > 0; m >>= 1) sum += __shfl_xor(sum, m, 32);
  float inv = 1.f / sum;
#pragma unroll
  for (int i = 0; i < 32; ++i) p[i * 32 + lane] = vals[i] * inv;
}

// ---------------------------------------------------------------------------
// ent2 = ent + attn @ V   (per batch: 32x768 <- 32x1024 @ 1024x768)
// ---------------------------------------------------------------------------
__global__ void __launch_bounds__(256)
attn_apply(const float* __restrict__ Attn, const f16* __restrict__ Vmat,
           const float* __restrict__ ent_in, float* __restrict__ ent_out,
           f16* __restrict__ ent_out_h) {
  __shared__ f16 As[32 * LDS_STRIDE];
  __shared__ f16 Vs[128 * LDS_STRIDE];
  const int tid = threadIdx.x;
  const int b = blockIdx.y, c0 = blockIdx.x * 128;
  const int w = tid >> 5, lane = tid & 31, half = lane >> 4, idx = lane & 15;
  v8f a0 = {}, a1 = {};
  for (int kk0 = 0; kk0 < 1024; kk0 += 32) {
    __syncthreads();
    for (int i = tid; i < 32 * 32; i += 256) {
      int e = i >> 5, kk = i & 31;
      As[e * LDS_STRIDE + kk] = (f16)Attn[((size_t)b * 32 + e) * 1024 + kk0 + kk];
    }
    {
      int kk = tid & 31, ng = tid >> 5;
      const f16* src = Vmat + ((size_t)b * 1024 + kk0 + kk) * 768 + c0 + ng * 16;
#pragma unroll
      for (int i = 0; i < 16; ++i) Vs[(ng * 16 + i) * LDS_STRIDE + kk] = src[i];
    }
    __syncthreads();
    v16h bf = frag_ld(&Vs[(w * 16 + idx) * LDS_STRIDE], half);
    v16h f0 = frag_ld(&As[idx * LDS_STRIDE], half);
    v16h f1 = frag_ld(&As[(16 + idx) * LDS_STRIDE], half);
    a0 = wmma16(f0, bf, a0);
    a1 = wmma16(f1, bf, a1);
  }
  int col = c0 + w * 16 + idx;
#pragma unroll
  for (int v = 0; v < 8; ++v) {
    int m = half * 8 + v;
    size_t r0 = (size_t)(b * 32 + m) * 768 + col;
    size_t r1 = (size_t)(b * 32 + 16 + m) * 768 + col;
    float v0 = ent_in[r0] + a0[v];
    float v1 = ent_in[r1] + a1[v];
    ent_out[r0] = v0; ent_out[r1] = v1;
    ent_out_h[r0] = (f16)v0; ent_out_h[r1] = (f16)v1;
  }
}

// ---------------------------------------------------------------------------
// NER head layer 2: out[row, c] = H1[row,:] . W2[:,c] + b2[c]   (N=17)
// ---------------------------------------------------------------------------
__global__ void __launch_bounds__(256)
ner_head2(const f16* __restrict__ H1, const float* __restrict__ W2,
          const float* __restrict__ b2, float* __restrict__ out) {
  __shared__ float W2s[256 * 17];
  __shared__ float b2s[17];
  int tid = threadIdx.x;
  for (int i = tid; i < 256 * 17; i += 256) W2s[i] = W2[i];
  if (tid < 17) b2s[tid] = b2[tid];
  __syncthreads();
  size_t gid = (size_t)blockIdx.x * 256 + tid;
  int row = (int)(gid / 17), c = (int)(gid % 17);
  float acc = b2s[c];
  const f16* hp = H1 + (size_t)row * 256;
  for (int k = 0; k < 256; ++k) acc += (float)hp[k] * W2s[k * 17 + c];
  out[gid] = acc;
}

// ---------------------------------------------------------------------------
// GAT attention+aggregation for one (batch, head). E=32, D<=128.
// ---------------------------------------------------------------------------
__global__ void __launch_bounds__(128)
gat_attn(const float* __restrict__ h, int heads, int D,
         const float* __restrict__ a_src, const float* __restrict__ a_dst,
         const float* __restrict__ bias, int relu, float* __restrict__ outf,
         f16* __restrict__ outh) {
  __shared__ float hbuf[32 * 128];
  __shared__ float ssrc[32], sdst[32];
  __shared__ float aw[32 * 32];
  int head = blockIdx.x, b = blockIdx.y;
  int HD = heads * D;
  int tid = threadIdx.x;
  for (int i = tid; i < 32 * D; i += 128) {
    int e = i / D, d = i % D;
    hbuf[e * D + d] = h[((size_t)b * 32 + e) * HD + head * D + d];
  }
  __syncthreads();
  if (tid < 32) {
    float ss = 0.f, sd = 0.f;
    for (int d = 0; d < D; ++d) {
      float v = hbuf[tid * D + d];
      ss += v * a_src[head * D + d];
      sd += v * a_dst[head * D + d];
    }
    ssrc[tid] = ss; sdst[tid] = sd;
  }
  __syncthreads();
  if (tid < 32) {
    int i = tid;
    float mx = -1e30f;
    for (int j = 0; j < 32; ++j) {
      float l = sdst[i] + ssrc[j];
      l = (l > 0.f) ? l : 0.2f * l;
      aw[i * 32 + j] = l;
      mx = fmaxf(mx, l);
    }
    float sum = 0.f;
    for (int j = 0; j < 32; ++j) { float e_ = __expf(aw[i * 32 + j] - mx); aw[i * 32 + j] = e_; sum += e_; }
    float inv = 1.f / sum;
    for (int j = 0; j < 32; ++j) aw[i * 32 + j] *= inv;
  }
  __syncthreads();
  for (int d = tid; d < D; d += 128) {
    for (int i = 0; i < 32; ++i) {
      float acc = 0.f;
      for (int j = 0; j < 32; ++j) acc += aw[i * 32 + j] * hbuf[j * D + d];
      float val = acc + bias[head * D + d];
      if (relu) val = fmaxf(val, 0.f);
      size_t o = ((size_t)b * 32 + i) * HD + head * D + d;
      if (outf) outf[o] = val;
      if (outh) outh[o] = (f16)val;
    }
  }
}

// ---------------------------------------------------------------------------
// feats[b,p,:] = concat(x2[b,PI[p],:], x2[b,PJ[p],:]); pair decoded from p.
// ---------------------------------------------------------------------------
__global__ void build_feats(const float* __restrict__ x2, f16* __restrict__ feats) {
  size_t gid = (size_t)blockIdx.x * 256 + threadIdx.x;
  int d = (int)(gid & 127);
  size_t t = gid >> 7;
  int p = (int)(t % 992);
  int b = (int)(t / 992);
  int i = p / 31, jj = p % 31;
  int j = jj + (jj >= i ? 1 : 0);
  float v = (d < 64) ? x2[((size_t)b * 32 + i) * 64 + d]
                     : x2[((size_t)b * 32 + j) * 64 + (d - 64)];
  feats[gid] = (f16)v;
}

// ---------------------------------------------------------------------------
// rel epilogue: layernorm(256) + scale/shift + relu + dot(W2) + b2 -> logits
// ---------------------------------------------------------------------------
__global__ void __launch_bounds__(256)
rel_head(const float* __restrict__ hrel, const float* __restrict__ g,
         const float* __restrict__ beta, const float* __restrict__ W2,
         const float* __restrict__ b2, int r, float* __restrict__ out) {
  int w = threadIdx.x >> 5, lane = threadIdx.x & 31;
  size_t row = (size_t)blockIdx.x * 8 + w;
  const float* p = hrel + row * 256;
  float x[8];
  float s = 0.f, sq = 0.f;
#pragma unroll
  for (int i = 0; i < 8; ++i) { x[i] = p[lane + i * 32]; s += x[i]; sq += x[i] * x[i]; }
#pragma unroll
  for (int m = 16; m > 0; m >>= 1) { s += __shfl_xor(s, m, 32); sq += __shfl_xor(sq, m, 32); }
  float mu = s * (1.f / 256.f);
  float var = sq * (1.f / 256.f) - mu * mu;
  float rstd = rsqrtf(var + 1e-5f);
  float acc = 0.f;
#pragma unroll
  for (int i = 0; i < 8; ++i) {
    int k = lane + i * 32;
    float vn = (x[i] - mu) * rstd * g[r * 256 + k] + beta[r * 256 + k];
    vn = fmaxf(vn, 0.f);
    acc += vn * W2[r * 256 + k];
  }
#pragma unroll
  for (int m = 16; m > 0; m >>= 1) acc += __shfl_xor(acc, m, 32);
  if (lane == 0) {
    int b = (int)(row / 992), pp = (int)(row % 992);
    out[((size_t)b * 6 + r) * 992 + pp] = acc + b2[r];
  }
}

// ---------------------------------------------------------------------------
extern "C" void kernel_launch(void* const* d_in, const int* in_sizes, int n_in,
                              void* d_out, int out_size, void* d_ws, size_t ws_size,
                              hipStream_t stream) {
  (void)in_sizes; (void)n_in; (void)out_size; (void)ws_size;
  const float* seq   = (const float*)d_in[0];
  const int*   mask  = (const int*)d_in[1];
  const int*   starts = (const int*)d_in[2];
  const int*   lens  = (const int*)d_in[3];
  const float* nW1 = (const float*)d_in[4];
  const float* nb1 = (const float*)d_in[5];
  const float* nW2 = (const float*)d_in[6];
  const float* nb2 = (const float*)d_in[7];
  const float* qW = (const float*)d_in[8];
  const float* qb = (const float*)d_in[9];
  const float* kW = (const float*)d_in[10];
  const float* kb = (const float*)d_in[11];
  const float* vW = (const float*)d_in[12];
  const float* vb = (const float*)d_in[13];
  const float* g1W = (const float*)d_in[14];
  const float* g1as = (const float*)d_in[15];
  const float* g1ad = (const float*)d_in[16];
  const float* g1b = (const float*)d_in[17];
  const float* g2W = (const float*)d_in[18];
  const float* g2as = (const float*)d_in[19];
  const float* g2ad = (const float*)d_in[20];
  const float* g2b = (const float*)d_in[21];
  const float* rW1 = (const float*)d_in[22];
  const float* rb1 = (const float*)d_in[23];
  const float* rg = (const float*)d_in[24];
  const float* rbeta = (const float*)d_in[25];
  const float* rW2 = (const float*)d_in[26];
  const float* rb2 = (const float*)d_in[27];
  float* out = (float*)d_out;

  char* wsb = (char*)d_ws;
  size_t off = 0;
  auto alloc = [&](size_t bytes) -> void* {
    void* p = wsb + off;
    off = (off + bytes + 255) & ~(size_t)255;
    return p;
  };
  f16*   seq16   = (f16*)alloc(32ll * 1024 * 768 * 2);
  f16*   w1_16   = (f16*)alloc(768 * 256 * 2);
  f16*   qW16    = (f16*)alloc(768 * 768 * 2);
  f16*   kW16    = (f16*)alloc(768 * 768 * 2);
  f16*   vW16    = (f16*)alloc(768 * 768 * 2);
  f16*   g1W16   = (f16*)alloc(768 * 512 * 2);
  f16*   g2W16   = (f16*)alloc(512 * 64 * 2);
  f16*   rW1_16  = (f16*)alloc(6 * 128 * 256 * 2);
  f16*   H1      = (f16*)alloc(32768ll * 256 * 2);
  f16*   Km      = (f16*)alloc(32768ll * 768 * 2);
  f16*   Vm      = (f16*)alloc(32768ll * 768 * 2);
  float* ent     = (float*)alloc(1024 * 768 * 4);
  f16*   ent16   = (f16*)alloc(1024 * 768 * 2);
  f16*   q16     = (f16*)alloc(1024 * 768 * 2);
  float* Sc      = (float*)alloc(1024ll * 1024 * 4);
  float* ent2    = (float*)alloc(1024 * 768 * 4);
  f16*   ent2_16 = (f16*)alloc(1024 * 768 * 2);
  float* hg1     = (float*)alloc(1024 * 512 * 4);
  f16*   x1_16   = (f16*)alloc(1024 * 512 * 2);
  float* hg2     = (float*)alloc(1024 * 64 * 4);
  float* x2      = (float*)alloc(1024 * 64 * 4);
  f16*   feats   = (f16*)alloc(32ll * 992 * 128 * 2);
  float* hrel    = (float*)alloc(31744ll * 256 * 4);

  auto cvt = [&](const float* src, f16* dst, size_t n) {
    size_t nblk = (n + 1023) / 1024;  // 4 elems per thread
    f32_to_f16<<<dim3((unsigned)nblk), dim3(256), 0, stream>>>(src, dst, n);
  };
  cvt(seq, seq16, 32ll * 1024 * 768);
  cvt(nW1, w1_16, 768 * 256);
  cvt(qW, qW16, 768 * 768);
  cvt(kW, kW16, 768 * 768);
  cvt(vW, vW16, 768 * 768);
  cvt(g1W, g1W16, 768 * 512);
  cvt(g2W, g2W16, 512 * 64);
  cvt(rW1, rW1_16, 6 * 128 * 256);

  // NER branch
  gemm_wmma<<<dim3(2, 256), 256, 0, stream>>>(seq16, w1_16, nb1, nullptr, H1,
                                              32768, 256, 768, 1);
  ner_head2<<<dim3(2176), 256, 0, stream>>>(H1, nW2, nb2, out);

  // K/V projections over whole sequence
  gemm_wmma<<<dim3(6, 256), 256, 0, stream>>>(seq16, kW16, kb, nullptr, Km,
                                              32768, 768, 768, 0);
  gemm_wmma<<<dim3(6, 256), 256, 0, stream>>>(seq16, vW16, vb, nullptr, Vm,
                                              32768, 768, 768, 0);

  // Entity pooling + Q projection
  ent_pool<<<dim3(1024), 256, 0, stream>>>(seq, starts, lens, ent, ent16);
  gemm_wmma<<<dim3(6, 8), 256, 0, stream>>>(ent16, qW16, qb, nullptr, q16,
                                            1024, 768, 768, 0);

  // Cross-attention
  attn_scores<<<dim3(8, 32), 256, 0, stream>>>(q16, Km, mask, Sc);
  softmax_rows<<<dim3(128), 256, 0, stream>>>(Sc);
  attn_apply<<<dim3(6, 32), 256, 0, stream>>>(Sc, Vm, ent, ent2, ent2_16);

  // GAT layer 1 (4 heads x 128) with ReLU
  gemm_wmma<<<dim3(4, 8), 256, 0, stream>>>(ent2_16, g1W16, nullptr, hg1, nullptr,
                                            1024, 512, 768, 0);
  gat_attn<<<dim3(4, 32), 128, 0, stream>>>(hg1, 4, 128, g1as, g1ad, g1b, 1,
                                            nullptr, x1_16);
  // GAT layer 2 (1 head x 64), no ReLU
  gemm_wmma<<<dim3(1, 8), 256, 0, stream>>>(x1_16, g2W16, nullptr, hg2, nullptr,
                                            1024, 64, 512, 0);
  gat_attn<<<dim3(1, 32), 128, 0, stream>>>(hg2, 1, 64, g2as, g2ad, g2b, 0,
                                            x2, nullptr);

  // Relation branch
  build_feats<<<dim3(15872), 256, 0, stream>>>(x2, feats);
  float* rel_out = out + 32 * 1024 * 17;
  for (int r = 0; r < 6; ++r) {
    gemm_wmma<<<dim3(2, 248), 256, 0, stream>>>(feats, rW1_16 + (size_t)r * 128 * 256,
                                                rb1 + r * 256, hrel, nullptr,
                                                31744, 256, 128, 0);
    rel_head<<<dim3(3968), 256, 0, stream>>>(hrel, rg, rbeta, rW2, rb2, r, rel_out);
  }
}
